// WaterNet0110_16690242913230
// MI455X (gfx1250) — compile-verified, matrix operand order checked by reading, not run on hardware
//
#include <hip/hip_runtime.h>

// ---------------------------------------------------------------------------
// WaterNet on MI455X (gfx1250, wave32, WMMA).
// Dominant cost: fcT MLP over 584,000 rows (38 -> 256 tanh -> 384) = 126 GFLOP
//   -> f16 WMMA (v_wmma_f32_16x16x32_f16), fp32 accumulate.
// Weights are pre-swizzled (one-shot kernel) into per-lane fragment-major
// order so every B-fragment is ONE contiguous 32B LDS read (2x ds_load_b128)
// and LDS staging is pure b128 copies. A/H fragments are hoisted and reused
// across all N-tiles. Epilogues fuse Pl/Ev/Vm so v is never materialized.
// Activations use v_exp_f32 + v_rcp_f32 (no IEEE divide sequences).
// ---------------------------------------------------------------------------

typedef __attribute__((ext_vector_type(16))) _Float16 v16h;
typedef __attribute__((ext_vector_type(8)))  _Float16 v8h;
typedef __attribute__((ext_vector_type(8)))  float    v8f;

constexpr int NH = 128, NG = 32, NR = 15;
constexpr int NT = 730, NS = 800;
constexpr long M = (long)NT * NS;            // 584,000 rows
constexpr int KIN = 38;                      // 6 forcings + 32 attrs (pad to 64)

constexpr int W1S_HALVES = 16 * 2 * 32 * 16;       // 16384  (16 nt x 2 kt)
constexpr int W2S_HALVES = 3 * 8 * 8 * 32 * 16;    // 98304  (3 chunks x 8 nt x 8 kt)

// dynamic-LDS layout for the WMMA MLP kernel (178.5 KB; CDNA5 allows 320 KB/WG)
constexpr int S3_A1 = 0;                       // 128 x 64  f16 (row-major)
constexpr int S3_W1 = S3_A1 + 128 * 64 * 2;    // swizzled W1, 32 KB
constexpr int S3_H  = S3_W1 + W1S_HALVES * 2;  // 128 x 256 f16 (row-major tanh hidden)
constexpr int S3_W2 = S3_H  + 128 * 256 * 2;   // swizzled W2 chunk, 64 KB
constexpr int S3_B1 = S3_W2 + 8 * 8 * 32 * 16 * 2;
constexpr int S3_B2 = S3_B1 + 256 * 4;
constexpr int S3_SZ = S3_B2 + 384 * 4;         // = 182,784 bytes

// fast sigmoid / tanh built on v_exp_f32 + v_rcp_f32 (no IEEE div sequence)
__device__ __forceinline__ float sigf(float a) {
  return __builtin_amdgcn_rcpf(1.f + __expf(-a));
}
// tanh(x) = 1 - 2/(exp(2x)+1); exact at +/-inf, ~1 ulp of rcp elsewhere —
// plenty for an f16-quantized hidden state.
__device__ __forceinline__ float fast_tanh(float a) {
  const float e = __expf(2.f * a);
  return 1.f - 2.f * __builtin_amdgcn_rcpf(e + 1.f);
}
// inverse of the ISA 7.12.2 16-bit fragment mapping k = e + (e&8) + 8*hsel
__device__ __forceinline__ int frag_k(int e, int hsel) { return e + (e & 8) + 8 * hsel; }

// ---------------------------------------------------------------------------
// Kernel 0: one-shot weight pre-swizzle (f32 -> f16, fragment-major).
// dest layout: [nt][kt][lane][e], lane = hsel*16 + n_in_tile.
// ---------------------------------------------------------------------------
__global__ void __launch_bounds__(256) wn_prep_w(
    const float* __restrict__ w1, const float* __restrict__ w2,
    _Float16* __restrict__ gW1s, _Float16* __restrict__ gW2s)
{
  const int i = blockIdx.x * 256 + threadIdx.x;
  if (i < W1S_HALVES) {
    const int e = i & 15, lane = (i >> 4) & 31, kt = (i >> 9) & 1, nt = i >> 10;
    const int n = nt * 16 + (lane & 15);
    const int k = kt * 32 + frag_k(e, lane >> 4);
    gW1s[i] = (_Float16)((k < KIN) ? w1[k * 256 + n] : 0.f);
  }
  const int j = i - W1S_HALVES;
  if (j >= 0 && j < W2S_HALVES) {
    const int e = j & 15, lane = (j >> 4) & 31, kt = (j >> 9) & 7;
    const int nt = (j >> 12) & 7, c = j >> 15;
    const int n = c * 128 + nt * 16 + (lane & 15);
    const int k = kt * 32 + frag_k(e, lane >> 4);
    gW2s[j] = (_Float16)w2[k * 384 + n];
  }
}

// ---------------------------------------------------------------------------
// Kernel 1: static parameter heads (fcW, fcR) per site. 800 WGs x 256 thr.
// ---------------------------------------------------------------------------
__global__ void __launch_bounds__(256) wn_static_heads(
    const float* __restrict__ xc,
    const float* __restrict__ w1W, const float* __restrict__ b1W,
    const float* __restrict__ w2W, const float* __restrict__ b2W,
    const float* __restrict__ w1R, const float* __restrict__ b1R,
    const float* __restrict__ w2R, const float* __restrict__ b2R,
    float* __restrict__ kp, float* __restrict__ ksx, float* __restrict__ kg,
    float* __restrict__ gp, float* __restrict__ gL, float* __restrict__ qb,
    float* __restrict__ ga, float* __restrict__ r)
{
  __shared__ float sxc[NG];
  __shared__ float sHW[256];
  __shared__ float sHR[256];
  __shared__ float sga[NH];
  __shared__ float sred[NH];

  const int s = blockIdx.x, tid = threadIdx.x;
  if (tid < NG) sxc[tid] = xc[s * NG + tid];
  __syncthreads();

  {
    float aW = b1W[tid], aR = b1R[tid];
    #pragma unroll 8
    for (int k = 0; k < NG; ++k) {
      const float xv = sxc[k];
      aW += xv * w1W[k * 256 + tid];
      aR += xv * w1R[k * 256 + tid];
    }
    sHW[tid] = fast_tanh(aW);
    sHR[tid] = fast_tanh(aR);
  }
  __syncthreads();

  for (int o = tid; o < NH * 7; o += 256) {
    float a = b2W[o];
    for (int k = 0; k < 256; ++k) a += sHW[k] * w2W[k * (NH * 7) + o];
    const int blk = o >> 7, h = o & 127;
    const long idx = (long)s * NH + h;
    switch (blk) {
      case 0: kp[idx]  = sigf(a);            break;
      case 1: ksx[idx] = sigf(a);            break;
      case 2: kg[idx]  = sigf(a) * 0.1f;     break;
      case 3: gp[idx]  = sigf(a);            break;
      case 4: gL[idx]  = __expf(a) * 2.f;    break;
      case 5: qb[idx]  = fmaxf(a, 0.f);      break;
      default: sga[h]  = a;                  break;
    }
  }
  for (int o = tid; o < NH * NR; o += 256) {
    float a = b2R[o];
    for (int k = 0; k < 256; ++k) a += sHR[k] * w2R[k * (NH * NR) + o];
    r[(long)s * NH * NR + o] = fmaxf(a, 0.f);
  }
  __syncthreads();

  if (tid < NH) sred[tid] = sga[tid];
  __syncthreads();
  for (int off = 64; off > 0; off >>= 1) {
    if (tid < off) sred[tid] = fmaxf(sred[tid], sred[tid + off]);
    __syncthreads();
  }
  const float mx = sred[0];
  __syncthreads();
  if (tid < NH) { const float e = __expf(sga[tid] - mx); sga[tid] = e; sred[tid] = e; }
  __syncthreads();
  for (int off = 64; off > 0; off >>= 1) {
    if (tid < off) sred[tid] += sred[tid + off];
    __syncthreads();
  }
  const float sm = sred[0];
  if (tid < NH) ga[(long)s * NH + tid] = sga[tid] * __builtin_amdgcn_rcpf(sm);
}

// ---------------------------------------------------------------------------
// Kernel 2: snow fraction, Ps = P*vf, Pliq = P*(1-vf).
// ---------------------------------------------------------------------------
__global__ void __launch_bounds__(256) wn_snow(
    const float* __restrict__ x, float* __restrict__ Ps, float* __restrict__ Pliq)
{
  const long i = (long)blockIdx.x * 256 + threadIdx.x;
  if (i >= M) return;
  const float P  = x[i * 6 + 0];
  const float T1 = x[i * 6 + 2];
  const float T2 = x[i * 6 + 3];
  const float den = T2 - T1;
  float cf = (T1 + T2) / ((den == 0.f) ? 1.f : den);
  cf = fminf(fmaxf(cf, -0.999999f), 0.999999f);
  float vf = acosf(cf) * (1.f / 3.1415f);
  if (T1 >= 0.f) vf = 0.f;
  if (T2 <= 0.f) vf = 1.f;
  Ps[i]   = P * vf;
  Pliq[i] = P * (1.f - vf);
}

// ---------------------------------------------------------------------------
// Kernel 3: fcT MLP with WMMA. 128 rows/WG (8 waves x 16-row strips).
// Layer1: 16 nt x 2 kt = 32 WMMA/wave.  Layer2: 3 chunks x 8 nt x 8 kt =
// 192 WMMA/wave with the 8 H fragments held in VGPRs across all tiles.
// Every B fragment: one contiguous 32B LDS read from the swizzled buffer.
// ---------------------------------------------------------------------------
__global__ void __launch_bounds__(256) wn_mlpT_wmma(
    const float* __restrict__ x,  const float* __restrict__ xc,
    const _Float16* __restrict__ gW1s, const _Float16* __restrict__ gW2s,
    const float* __restrict__ b1, const float* __restrict__ b2,
    const float* __restrict__ Pliq,
    float* __restrict__ Pl, float* __restrict__ Ev, float* __restrict__ Vm)
{
  extern __shared__ char smem[];
  _Float16* sA   = (_Float16*)(smem + S3_A1);
  _Float16* sW1s = (_Float16*)(smem + S3_W1);
  _Float16* sH   = (_Float16*)(smem + S3_H);
  _Float16* sW2s = (_Float16*)(smem + S3_W2);
  float*    sb1  = (float*)(smem + S3_B1);
  float*    sb2  = (float*)(smem + S3_B2);

  const int tid  = threadIdx.x;
  const int lane = tid & 31, wv = tid >> 5;
  const int nl   = lane & 15;        // A-row m / B-col n within a tile
  const int hsel = lane >> 4;        // lane-half select
  const long row0 = (long)blockIdx.x * 128;

  // stage A rows (f32 -> f16, zero-pad K 38..63 and rows past M)
  for (int i = tid; i < 128 * 64; i += 256) {
    const int rr = i >> 6, k = i & 63;
    const long grow = row0 + rr;
    float v = 0.f;
    if (grow < M) {
      if (k < 6)        v = x[grow * 6 + k];
      else if (k < KIN) v = xc[(grow % NS) * NG + (k - 6)];
    }
    sA[i] = (_Float16)v;
  }
  // stage swizzled W1: pure contiguous 16B copies
  {
    const uint4* src = (const uint4*)gW1s;
    uint4* dst = (uint4*)sW1s;
    #pragma unroll
    for (int i = 0; i < W1S_HALVES / 8 / 256; ++i)
      dst[tid + i * 256] = src[tid + i * 256];
  }
  if (tid < 256) sb1[tid] = b1[tid];
  for (int i = tid; i < 384; i += 256) sb2[i] = b2[i];
  __syncthreads();

  // ---- layer 1: H = tanh(A @ W1 + b1), stored f16 row-major in LDS ----
  v16h aF[2];
  {
    const int arow = (wv * 16 + nl) * 64;
    #pragma unroll
    for (int kt = 0; kt < 2; ++kt) {
      const v8h lo = *(const v8h*)(&sA[arow + kt * 32 + 8 * hsel]);
      const v8h hi = *(const v8h*)(&sA[arow + kt * 32 + 16 + 8 * hsel]);
      aF[kt] = __builtin_shufflevector(lo, hi, 0,1,2,3,4,5,6,7,8,9,10,11,12,13,14,15);
    }
  }
  #pragma unroll
  for (int nt = 0; nt < 16; ++nt) {
    v8f acc = {};
    #pragma unroll
    for (int kt = 0; kt < 2; ++kt) {
      const v16h bf = *(const v16h*)(&sW1s[((nt * 2 + kt) * 32 + lane) * 16]);
      acc = __builtin_amdgcn_wmma_f32_16x16x32_f16(false, aF[kt], false, bf,
                                                   (short)0, acc, false, false);
    }
    #pragma unroll
    for (int g = 0; g < 8; ++g) {
      const int m = g + 8 * hsel;
      const int col = nt * 16 + nl;
      sH[(wv * 16 + m) * 256 + col] = (_Float16)fast_tanh(acc[g] + sb1[col]);
    }
  }
  __syncthreads();

  // hoist the 8 H fragments of this wave's strip (reused by all 24 acc tiles)
  v16h hF[8];
  {
    const int hrow = (wv * 16 + nl) * 256;
    #pragma unroll
    for (int kt = 0; kt < 8; ++kt) {
      const v8h lo = *(const v8h*)(&sH[hrow + kt * 32 + 8 * hsel]);
      const v8h hi = *(const v8h*)(&sH[hrow + kt * 32 + 16 + 8 * hsel]);
      hF[kt] = __builtin_shufflevector(lo, hi, 0,1,2,3,4,5,6,7,8,9,10,11,12,13,14,15);
    }
  }

  // ---- layer 2: 3 column chunks of 128 == vi | ve | vm ----
  for (int c = 0; c < 3; ++c) {
    __syncthreads();                 // all waves done with previous chunk
    {
      const uint4* src = (const uint4*)(gW2s + (long)c * (8 * 8 * 32 * 16));
      uint4* dst = (uint4*)sW2s;
      #pragma unroll
      for (int i = 0; i < (8 * 8 * 32 * 16) / 8 / 256; ++i)
        dst[tid + i * 256] = src[tid + i * 256];
    }
    __syncthreads();

    #pragma unroll
    for (int nt = 0; nt < 8; ++nt) {
      v8f acc = {};
      #pragma unroll
      for (int kt = 0; kt < 8; ++kt) {
        const v16h bf = *(const v16h*)(&sW2s[((nt * 8 + kt) * 32 + lane) * 16]);
        acc = __builtin_amdgcn_wmma_f32_16x16x32_f16(false, hF[kt], false, bf,
                                                     (short)0, acc, false, false);
      }
      #pragma unroll
      for (int g = 0; g < 8; ++g) {
        const int m = g + 8 * hsel;
        const long grow = row0 + wv * 16 + m;
        if (grow >= M) continue;
        const int h = nt * 16 + nl;                    // bucket index 0..127
        const float v = acc[g] + sb2[c * 128 + h];
        const long o = grow * NH + h;
        if (c == 0) {
          const float vi = fminf(fmaxf(v * (1.f / 3.f) + 0.5f, 0.f), 1.f);
          Pl[o] = Pliq[grow] * vi;
        } else if (c == 1) {
          Ev[o] = x[grow * 6 + 1] * (fmaxf(v, 0.f) * 2.f);
        } else {
          Vm[o] = __expf(v);
        }
      }
    }
  }
}

// ---------------------------------------------------------------------------
// Kernel 4: sequential reservoir scan, bandwidth-bound.
// ---------------------------------------------------------------------------
__global__ void __launch_bounds__(128) wn_scan(
    const float* __restrict__ Ps, const float* __restrict__ Pl,
    const float* __restrict__ Ev, const float* __restrict__ Vm,
    const float* __restrict__ kp, const float* __restrict__ ksx,
    const float* __restrict__ kg, const float* __restrict__ gp,
    const float* __restrict__ gL, const float* __restrict__ qb,
    float* __restrict__ QT)
{
  const int s = blockIdx.x, h = threadIdx.x;
  const long pidx = (long)s * NH + h;
  const float Kp = kp[pidx], Ks = ksx[pidx], Kg = kg[pidx];
  const float Gp = gp[pidx], Gl = gL[pidx], Qb = qb[pidx];
  float Sf = 0.f, Ss = 0.f, Sg = 0.f;

  for (int t = 0; t < NT; ++t) {
    const long base = ((long)t * NS + s) * NH + h;
    if (t + 1 < NT) {                       // gfx1250 global_prefetch_b8
      const long nb = base + (long)NS * NH;
      __builtin_prefetch(&Pl[nb], 0, 0);
      __builtin_prefetch(&Ev[nb], 0, 0);
      __builtin_prefetch(&Vm[nb], 0, 0);
    }
    const float ps = Ps[t * NS + s];
    const float pl = Pl[base], ev = Ev[base], vmk = Vm[base];

    const float a  = Sf + ps;
    const float qf = fminf(a, vmk);
    Sf = fmaxf(a - vmk, 0.f);
    const float H  = fmaxf(Ss + pl + qf - ev, 0.f);
    const float qp = fmaxf(Kp * (H - Gl), 0.f);
    const float qs = Ks * fminf(H, Gl);
    Ss = H - qp - qs;
    const float qg = Kg * (Sg + qs * Gp) + Qb;
    Sg = Sg - qg;
    QT[base] = qp + qs * (1.f - Gp) + qg;
  }
}

// ---------------------------------------------------------------------------
// Kernel 5: 15-tap causal FIR + ga reduction, QT tiled through LDS.
// ---------------------------------------------------------------------------
__global__ void __launch_bounds__(256) wn_route(
    const float* __restrict__ QT, const float* __restrict__ r,
    const float* __restrict__ ga, float* __restrict__ out)
{
  constexpr int TT = 32;
  __shared__ float q[(TT + NR - 1) * NH];     // 46 x 128
  __shared__ float rb[NH * NR];
  __shared__ float gb[NH];
  __shared__ float part[TT][8];

  const int s = blockIdx.y, t0 = blockIdx.x * TT, tid = threadIdx.x;

  for (int i = tid; i < (TT + NR - 1) * NH; i += 256) {
    const int tt = i >> 7, h = i & 127;
    const int gt = t0 - (NR - 1) + tt;
    q[i] = (gt >= 0 && gt < NT) ? QT[((long)gt * NS + s) * NH + h] : 0.f;
  }
  for (int i = tid; i < NH * NR; i += 256) rb[i] = r[(long)s * NH * NR + i];
  if (tid < NH) gb[tid] = ga[(long)s * NH + tid];
  __syncthreads();

  const int tl = tid >> 3, hg = tid & 7;
  const int t = t0 + tl;
  float acc = 0.f;
  if (t < NT) {
    for (int h = hg * 16; h < hg * 16 + 16; ++h) {
      float fir = 0.f;
      #pragma unroll
      for (int i = 0; i < NR; ++i)
        fir += q[(tl + NR - 1 - i) * NH + h] * rb[h * NR + i];
      acc += gb[h] * fir;
    }
  }
  part[tl][hg] = acc;
  __syncthreads();
  if (hg == 0 && t < NT) {
    float sum = 0.f;
    #pragma unroll
    for (int j = 0; j < 8; ++j) sum += part[tl][j];
    out[(long)t * NS + s] = sum;
  }
}

// ---------------------------------------------------------------------------
extern "C" void kernel_launch(void* const* d_in, const int* in_sizes, int n_in,
                              void* d_out, int out_size, void* d_ws, size_t ws_size,
                              hipStream_t stream) {
  const float* x      = (const float*)d_in[0];
  const float* xc     = (const float*)d_in[1];
  const float* fcW1_w = (const float*)d_in[2];
  const float* fcW1_b = (const float*)d_in[3];
  const float* fcW2_w = (const float*)d_in[4];
  const float* fcW2_b = (const float*)d_in[5];
  const float* fcT1_w = (const float*)d_in[6];
  const float* fcT1_b = (const float*)d_in[7];
  const float* fcT2_w = (const float*)d_in[8];
  const float* fcT2_b = (const float*)d_in[9];
  const float* fcR1_w = (const float*)d_in[10];
  const float* fcR1_b = (const float*)d_in[11];
  const float* fcR2_w = (const float*)d_in[12];
  const float* fcR2_b = (const float*)d_in[13];
  float* out = (float*)d_out;

  // workspace carve-up (fp32 planes + f16 swizzled weights): ~1.21 GB total
  float* ws = (float*)d_ws;
  constexpr long PAR = (long)NS * NH;             // 102,400
  float* kp   = ws;
  float* ksx  = ws + 1 * PAR;
  float* kg   = ws + 2 * PAR;
  float* gp   = ws + 3 * PAR;
  float* gL   = ws + 4 * PAR;
  float* qb   = ws + 5 * PAR;
  float* ga   = ws + 6 * PAR;
  float* r    = ws + 7 * PAR;                     // NS*NH*NR
  float* Ps   = r    + (long)NS * NH * NR;
  float* Pliq = Ps   + M;
  float* Pl   = Pliq + M;
  float* Ev   = Pl   + M * NH;
  float* Vm   = Ev   + M * NH;
  float* QT   = Vm   + M * NH;
  _Float16* gW1s = (_Float16*)(QT + M * NH);      // 16-byte aligned offset
  _Float16* gW2s = gW1s + W1S_HALVES;
  const size_t need = (size_t)((float*)(gW2s + W2S_HALVES) - ws) * sizeof(float)
                      + 64;   // slack
  if (ws_size < need) return;

  wn_prep_w<<<(W1S_HALVES + W2S_HALVES + 255) / 256, 256, 0, stream>>>(
      fcT1_w, fcT2_w, gW1s, gW2s);

  wn_static_heads<<<NS, 256, 0, stream>>>(
      xc, fcW1_w, fcW1_b, fcW2_w, fcW2_b, fcR1_w, fcR1_b, fcR2_w, fcR2_b,
      kp, ksx, kg, gp, gL, qb, ga, r);

  wn_snow<<<(int)((M + 255) / 256), 256, 0, stream>>>(x, Ps, Pliq);

  wn_mlpT_wmma<<<(int)((M + 127) / 128), 256, S3_SZ, stream>>>(
      x, xc, gW1s, gW2s, fcT1_b, fcT2_b, Pliq, Pl, Ev, Vm);

  wn_scan<<<NS, NH, 0, stream>>>(Ps, Pl, Ev, Vm, kp, ksx, kg, gp, gL, qb, QT);

  wn_route<<<dim3((NT + 31) / 32, NS), 256, 0, stream>>>(QT, r, ga, out);
}
